// SVDUFGConv_47047071760998
// MI455X (gfx1250) — matrix operand order repaired and checked by reading, not compiled, not run on hardware
//
#include <hip/hip_runtime.h>

#define NROWS 8192
#define NF    128
#define LDS_STRIDE 40   // bf16 elems per column: 32 data + 8 pad = 80 B (bank-conflict-free)

typedef __attribute__((ext_vector_type(16))) __bf16 v16bf;
typedef __attribute__((ext_vector_type(8)))  __bf16 v8bf;
typedef __attribute__((ext_vector_type(8)))  float  v8f;
typedef int v4i __attribute__((vector_size(16)));   // matches builtin param type

#if __has_builtin(__builtin_amdgcn_global_load_async_to_lds_b128)
#define ASYNC_CP 1
#else
#define ASYNC_CP 0
#endif

#if __has_builtin(__builtin_amdgcn_s_wait_asynccnt)
#define WAIT_ASYNC() __builtin_amdgcn_s_wait_asynccnt(0)
#elif ASYNC_CP
#define WAIT_ASYNC() asm volatile("s_wait_asynccnt 0x0" ::: "memory")
#else
#define WAIT_ASYNC() ((void)0)
#endif

// ---------------------------------------------------------------------------
// Kernel 1: scale[n] = sum_j d[j][n]^2 * filt[j][n]    (JD = 3)
// ---------------------------------------------------------------------------
__global__ void svduf_scale_kernel(const float* __restrict__ d,
                                   const float* __restrict__ filt,
                                   float* __restrict__ scale) {
    int n = blockIdx.x * blockDim.x + threadIdx.x;
    if (n >= NROWS) return;
    float s = 0.f;
#pragma unroll
    for (int j = 0; j < 3; ++j) {
        float dv = d[j * NROWS + n];
        s = fmaf(dv * dv, filt[j * NROWS + n], s);
    }
    scale[n] = s;
}

// ---------------------------------------------------------------------------
// Kernel 2: xwT_{hi,lo}[c][n] = bf16_split( sum_f x[n][f] * W[f][c] )
// Stored transposed [col][row] so it is a contiguous WMMA B-operand stream.
// ---------------------------------------------------------------------------
__global__ void svduf_xw_kernel(const float* __restrict__ x,
                                const float* __restrict__ W,
                                __bf16* __restrict__ xwT_hi,
                                __bf16* __restrict__ xwT_lo) {
    __shared__ float xs[NF];
    const int n = blockIdx.x;
    const int c = threadIdx.x;
    xs[c] = x[(size_t)n * NF + c];
    __syncthreads();
    float acc = 0.f;
#pragma unroll 8
    for (int f = 0; f < NF; ++f)
        acc = fmaf(xs[f], W[f * NF + c], acc);
    __bf16 hi = (__bf16)acc;
    __bf16 lo = (__bf16)(acc - (float)hi);
    xwT_hi[(size_t)c * NROWS + n] = hi;
    xwT_lo[(size_t)c * NROWS + n] = lo;
}

// ---------------------------------------------------------------------------
// Cooperative B-panel fill: thread == column, copies 64 B (32 x bf16) per
// matrix into LDS.  Async path uses GLOBAL_LOAD_ASYNC_TO_LDS_B128 (ASYNCcnt).
// ---------------------------------------------------------------------------
__device__ __forceinline__ void svduf_fill_b(const __bf16* __restrict__ Bhi,
                                             const __bf16* __restrict__ Blo,
                                             __bf16* shi, __bf16* slo,
                                             int col, int k0) {
    const __bf16* ghi = Bhi + (size_t)col * NROWS + k0;
    const __bf16* glo = Blo + (size_t)col * NROWS + k0;
    __bf16* lhi = shi + col * LDS_STRIDE;
    __bf16* llo = slo + col * LDS_STRIDE;
#if ASYNC_CP
#pragma unroll
    for (int q = 0; q < 4; ++q) {
        __builtin_amdgcn_global_load_async_to_lds_b128(
            (v4i*)(ghi + q * 8), (v4i*)(lhi + q * 8), 0, 0);
        __builtin_amdgcn_global_load_async_to_lds_b128(
            (v4i*)(glo + q * 8), (v4i*)(llo + q * 8), 0, 0);
    }
#else
#pragma unroll
    for (int q = 0; q < 4; ++q) {
        *(v8bf*)(lhi + q * 8) = *(const v8bf*)(ghi + q * 8);
        *(v8bf*)(llo + q * 8) = *(const v8bf*)(glo + q * 8);
    }
#endif
}

// ---------------------------------------------------------------------------
// Kernels 3/4: C[8192 x 128] = A[8192 x 8192] @ B[8192 x 128]
//   A: f32 row-major streamed from HBM, split to bf16 hi/lo in registers.
//   B: bf16 hi/lo [col][k]; staged per-block in LDS, double-buffered, async.
//   bf16x3: acc += Ahi*Bhi + Ahi*Blo + Alo*Bhi  (f32 accumulate).
//   FIRST : epilogue applies per-row scale, emits bf16 hi/lo transposed.
//   !FIRST: epilogue adds per-col bias, emits f32 row-major (final output).
// Block = 4 waves (128 thr); wave owns 16 rows x 128 cols (8 acc tiles).
// ---------------------------------------------------------------------------
template <bool FIRST>
__global__ __launch_bounds__(128) void svduf_gemm_kernel(
    const float*  __restrict__ A,
    const __bf16* __restrict__ Bhi,
    const __bf16* __restrict__ Blo,
    const float*  __restrict__ rowscale,   // FIRST only
    const float*  __restrict__ bias,       // !FIRST only
    __bf16* __restrict__ Chi,              // FIRST out [128][8192]
    __bf16* __restrict__ Clo,              // FIRST out [128][8192]
    float*  __restrict__ Cout)             // !FIRST out [8192][128]
{
    // [buf][hi=0/lo=1][col*LDS_STRIDE + k] : 40960 B total
    __shared__ __bf16 sB[2][2][NF * LDS_STRIDE];

    const int tid  = threadIdx.x;          // 0..127 == fill column
    const int lane = tid & 31;
    const int wave = tid >> 5;
    const int m    = lane & 15;            // row within A tile / col within B tile
    const int h    = lane >> 4;            // lane half selects K sub-range
    const int rowTile = (blockIdx.x * 4 + wave) * 16;

    v8f acc[8];
#pragma unroll
    for (int t = 0; t < 8; ++t)
#pragma unroll
        for (int r = 0; r < 8; ++r) acc[t][r] = 0.f;

    const float* Arow = A + (size_t)(rowTile + m) * NROWS;

    // prologue: fill buffer 0 with k-panel 0
    svduf_fill_b(Bhi, Blo, sB[0][0], sB[0][1], tid, 0);

    for (int ks = 0; ks < NROWS / 32; ++ks) {
        const int k0  = ks * 32;
        const int cur = ks & 1;

        WAIT_ASYNC();        // my fills of buf[cur] are complete
        __syncthreads();     // everyone's fills done; prev compute on buf[!cur] done

        if (ks + 1 < NROWS / 32)   // overlap next panel fill with this compute
            svduf_fill_b(Bhi, Blo, sB[cur ^ 1][0], sB[cur ^ 1][1], tid, k0 + 32);

        __builtin_prefetch(Arow + k0 + 256, 0, 0);  // A-stream prefetch

        // A fragment: elems 0..7 -> K = k0 + 8h + 0..7 ; elems 8..15 -> +16
        v8f f0 = *(const v8f*)(Arow + k0 + h * 8);
        v8f f1 = *(const v8f*)(Arow + k0 + 16 + h * 8);
        v16bf ahi, alo;
#pragma unroll
        for (int e = 0; e < 8; ++e) {
            __bf16 h0 = (__bf16)f0[e];
            ahi[e]     = h0;
            alo[e]     = (__bf16)(f0[e] - (float)h0);
            __bf16 h1 = (__bf16)f1[e];
            ahi[e + 8] = h1;
            alo[e + 8] = (__bf16)(f1[e] - (float)h1);
        }

#pragma unroll
        for (int ct = 0; ct < 8; ++ct) {
            // B fragment from LDS: lane col = ct*16+m, elems 0..7 -> K = 8h + e ; elems 8..15 -> +16
            const __bf16* lph = &sB[cur][0][(ct * 16 + m) * LDS_STRIDE + h * 8];
            const __bf16* lpl = &sB[cur][1][(ct * 16 + m) * LDS_STRIDE + h * 8];
            v8bf bh0 = *(const v8bf*)lph;
            v8bf bh1 = *(const v8bf*)(lph + 16);
            v8bf bl0 = *(const v8bf*)lpl;
            v8bf bl1 = *(const v8bf*)(lpl + 16);
            v16bf bhi, blo;
#pragma unroll
            for (int e = 0; e < 8; ++e) {
                bhi[e] = bh0[e]; bhi[e + 8] = bh1[e];
                blo[e] = bl0[e]; blo[e + 8] = bl1[e];
            }
            acc[ct] = __builtin_amdgcn_wmma_f32_16x16x32_bf16(
                false, ahi, false, bhi, (short)0, acc[ct], false, false);
            acc[ct] = __builtin_amdgcn_wmma_f32_16x16x32_bf16(
                false, ahi, false, blo, (short)0, acc[ct], false, false);
            acc[ct] = __builtin_amdgcn_wmma_f32_16x16x32_bf16(
                false, alo, false, bhi, (short)0, acc[ct], false, false);
        }
    }

    if (FIRST) {
        // t[n][c] = scale[n] * acc ; split to bf16 hi/lo, store transposed.
        float s[8];
#pragma unroll
        for (int r = 0; r < 8; ++r) s[r] = rowscale[rowTile + h * 8 + r];
#pragma unroll
        for (int ct = 0; ct < 8; ++ct) {
            const int col = ct * 16 + m;
            v8bf oh, ol;
#pragma unroll
            for (int r = 0; r < 8; ++r) {
                float v = acc[ct][r] * s[r];
                __bf16 vh = (__bf16)v;
                oh[r] = vh;
                ol[r] = (__bf16)(v - (float)vh);
            }
            const size_t off = (size_t)col * NROWS + rowTile + h * 8;
            *(v8bf*)(Chi + off) = oh;   // 8 consecutive rows -> one 16B store
            *(v8bf*)(Clo + off) = ol;
        }
    } else {
        // out[n][c] = acc + bias[c], f32 row-major
#pragma unroll
        for (int ct = 0; ct < 8; ++ct) {
            const int col = ct * 16 + m;
            const float b = bias[col];
#pragma unroll
            for (int r = 0; r < 8; ++r)
                Cout[(size_t)(rowTile + h * 8 + r) * NF + col] = acc[ct][r] + b;
        }
    }
}

// ---------------------------------------------------------------------------
extern "C" void kernel_launch(void* const* d_in, const int* in_sizes, int n_in,
                              void* d_out, int out_size, void* d_ws, size_t ws_size,
                              hipStream_t stream) {
    const float* x    = (const float*)d_in[0];
    const float* dls  = (const float*)d_in[1];
    const float* U    = (const float*)d_in[2];
    const float* Vt   = (const float*)d_in[3];
    const float* W    = (const float*)d_in[4];
    const float* filt = (const float*)d_in[5];
    const float* bias = (const float*)d_in[6];
    float* out = (float*)d_out;

    // workspace carving (all offsets 32B-aligned):
    //   scale: 32 KB | xwT hi/lo: 2 MB each | tT hi/lo: 2 MB each  (~8.03 MB)
    char* ws = (char*)d_ws;
    float*  scale  = (float*)ws;
    __bf16* xwT_hi = (__bf16*)(ws + (32u << 10));
    __bf16* xwT_lo = (__bf16*)(ws + (32u << 10) + (2u << 20));
    __bf16* tT_hi  = (__bf16*)(ws + (32u << 10) + (4u << 20));
    __bf16* tT_lo  = (__bf16*)(ws + (32u << 10) + (6u << 20));

    svduf_scale_kernel<<<NROWS / 256, 256, 0, stream>>>(dls, filt, scale);
    svduf_xw_kernel<<<NROWS, NF, 0, stream>>>(x, W, xwT_hi, xwT_lo);
    svduf_gemm_kernel<true ><<<NROWS / 64, 128, 0, stream>>>(
        Vt, xwT_hi, xwT_lo, scale, nullptr, tT_hi, tT_lo, nullptr);
    svduf_gemm_kernel<false><<<NROWS / 64, 128, 0, stream>>>(
        U, tT_hi, tT_lo, nullptr, bias, nullptr, nullptr, out);
}
